// LocalMP_26834955666029
// MI455X (gfx1250) — matrix operand order repaired and checked by reading, not compiled, not run on hardware
//
#include <hip/hip_runtime.h>
#include <hip/hip_bf16.h>

// Problem constants (match reference setup_inputs)
#define NATOMS 50000
#define NEDGES 600000
#define NCLIQ  20000
#define NPAIR  60000
#define NGRAPH 1024
#define HDIM   128
#define H2DIM  256
#define NLAYER 3
#define VA_SZ  128
#define VB_SZ  16
#define BN_EPS 1e-5f

typedef __attribute__((ext_vector_type(16))) _Float16 v16h;
typedef __attribute__((ext_vector_type(8)))  _Float16 v8h;
typedef __attribute__((ext_vector_type(8)))  float    v8f;
typedef __attribute__((ext_vector_type(4)))  float    v4f;

__device__ __forceinline__ void atomAddF(float* p, float v) {
  __hip_atomic_fetch_add(p, v, __ATOMIC_RELAXED, __HIP_MEMORY_SCOPE_AGENT);
}

// ---------------------------------------------------------------------------
// WMMA GEMM: C[M,Nc] = A[M,K](f32, cvt->f16 in-register) * WT[Nc,K](f16) + bias
// Each wave computes a 16x64 output strip (4 WMMA tiles) so the A fragment is
// loaded once per k-step and reused 4x. K multiple of 32; M mult of 16;
// Nc multiple of 64.
// ---------------------------------------------------------------------------
#define GN 4
__global__ __launch_bounds__(128)
void k_gemm_wmma(const float* __restrict__ A, const _Float16* __restrict__ WT,
                 const float* __restrict__ bias, float* __restrict__ C,
                 int M, int K, int Nc) {
  const int lane    = threadIdx.x & 31;
  const int waveId  = threadIdx.x >> 5;
  const int groupsN = Nc >> 6;                     // 64-wide column groups
  const int tilesM  = M >> 4;
  const int tile    = blockIdx.x * 4 + waveId;     // wave-uniform
  if (tile >= tilesM * groupsN) return;            // whole-wave exit: EXEC all-ones inside
  const int tm   = tile / groupsN;
  const int tg   = tile - tm * groupsN;
  const int half = lane >> 4;
  const int l16  = lane & 15;
  const int row  = tm * 16 + l16;                  // A row owned by this lane

  v8f acc[GN] = {};
  const float*    arow  = A  + (size_t)row * K;
  const _Float16* bbase = WT + (size_t)(tg * 64 + l16) * K;   // sub-tile s adds 16*K

  for (int k0 = 0; k0 < K; k0 += 32) {
    // A fragment: lanes 0-15 hold K = k0+0..7 / k0+16..23 ; lanes 16-31 hold +8
    v4f al0 = *(const v4f*)(arow + k0 + half * 8);
    v4f al1 = *(const v4f*)(arow + k0 + half * 8 + 4);
    v4f ah0 = *(const v4f*)(arow + k0 + 16 + half * 8);
    v4f ah1 = *(const v4f*)(arow + k0 + 16 + half * 8 + 4);
    v16h a;
#pragma unroll
    for (int j = 0; j < 4; ++j) {
      a[j]      = (_Float16)al0[j];
      a[4 + j]  = (_Float16)al1[j];
      a[8 + j]  = (_Float16)ah0[j];
      a[12 + j] = (_Float16)ah1[j];
    }
#pragma unroll
    for (int s = 0; s < GN; ++s) {
      // B fragment: lanes 0-15 hold K = k0+0..7 / k0+16..23 ; lanes 16-31 hold +8
      const _Float16* bp = bbase + (size_t)s * 16 * K + k0 + half * 8;
      v8h b0 = *(const v8h*)(bp);
      v8h b1 = *(const v8h*)(bp + 16);
      v16h b;
#pragma unroll
      for (int j = 0; j < 8; ++j) { b[j] = b0[j]; b[8 + j] = b1[j]; }
      acc[s] = __builtin_amdgcn_wmma_f32_16x16x32_f16(
          /*neg_a=*/false, a, /*neg_b=*/false, b,
          /*c_mod=*/(short)0, acc[s], /*reuse_a=*/false, /*reuse_b=*/false);
    }
  }

#pragma unroll
  for (int s = 0; s < GN; ++s) {
    const int col = tg * 64 + s * 16 + l16;
    const float bc = bias ? bias[col] : 0.0f;
#pragma unroll
    for (int v = 0; v < 8; ++v) {
      const int r = tm * 16 + half * 8 + v;        // C/D lane layout
      C[(size_t)r * Nc + col] = acc[s][v] + bc;
    }
  }
}

// ---------------------------------------------------------------------------
// Weight transpose + f32 -> f16 convert:  WT[n*K + k] = W[k*Nc + n]
// ---------------------------------------------------------------------------
__global__ __launch_bounds__(256)
void k_wT(const float* __restrict__ W, _Float16* __restrict__ WT, int K, int Nc) {
  long i = (long)blockIdx.x * blockDim.x + threadIdx.x;
  if (i >= (long)K * Nc) return;
  int k = (int)(i / Nc);
  int n = (int)(i - (long)k * Nc);
  WT[(size_t)n * K + k] = (_Float16)W[i];
}

__global__ __launch_bounds__(256)
void k_fill0(float* __restrict__ p, long n) {
  long i = (long)blockIdx.x * blockDim.x + threadIdx.x;
  if (i < n) p[i] = 0.0f;
}

__global__ __launch_bounds__(256)
void k_copy(const float* __restrict__ a, float* __restrict__ b, long n) {
  long i = (long)blockIdx.x * blockDim.x + threadIdx.x;
  if (i < n) b[i] = a[i];
}

// AtomEncoder: h[n,c] = sum_f atom_emb[f, x[n,f], c]; also graph_emb output
__global__ __launch_bounds__(256)
void k_atom_enc(const int* __restrict__ x, const float* __restrict__ aemb,
                float* __restrict__ h, float* __restrict__ gemb) {
  long i = (long)blockIdx.x * blockDim.x + threadIdx.x;
  if (i >= (long)NATOMS * HDIM) return;
  int n = (int)(i >> 7), c = (int)(i & 127);
  float s = 0.0f;
#pragma unroll
  for (int f = 0; f < 9; ++f) {
    int v = x[n * 9 + f];
    s += aemb[((size_t)f * VA_SZ + v) * HDIM + c];
  }
  h[i] = s;
  gemb[i] = s;
}

// Clique embedding gather (GEMM input): A[i,c] = clique_emb[x_clique[i], c]
__global__ __launch_bounds__(256)
void k_clq_gather(const int* __restrict__ xc_idx, const float* __restrict__ cemb,
                  float* __restrict__ A) {
  long i = (long)blockIdx.x * blockDim.x + threadIdx.x;
  if (i >= (long)NCLIQ * HDIM) return;
  int n = (int)(i >> 7), c = (int)(i & 127);
  A[i] = cemb[(size_t)xc_idx[n] * HDIM + c];
}

// Edge message + scatter: agg[dst[e],c] += relu(h[src[e],c] + bond_emb_sum)
__global__ __launch_bounds__(256)
void k_edge(const float* __restrict__ h, const float* __restrict__ bemb,
            const int* __restrict__ eattr, const int* __restrict__ src,
            const int* __restrict__ dst, float* __restrict__ agg) {
  long i = (long)blockIdx.x * blockDim.x + threadIdx.x;
  if (i >= (long)NEDGES * HDIM) return;
  int e = (int)(i >> 7), c = (int)(i & 127);
  float ea = 0.0f;
#pragma unroll
  for (int f = 0; f < 3; ++f) {
    int v = eattr[e * 3 + f];
    ea += bemb[((size_t)f * VB_SZ + v) * HDIM + c];
  }
  float m = h[(size_t)src[e] * HDIM + c] + ea;
  m = m > 0.0f ? m : 0.0f;
  atomAddF(&agg[(size_t)dst[e] * HDIM + c], m);
}

// z = (1+eps)*h + agg  (in place over agg buffer)
__global__ __launch_bounds__(256)
void k_make_z(const float* __restrict__ h, float* __restrict__ zb,
              const float* __restrict__ eps_arr, int layer, long n) {
  long i = (long)blockIdx.x * blockDim.x + threadIdx.x;
  if (i >= n) return;
  float eps = eps_arr[layer];
  zb[i] = (1.0f + eps) * h[i] + zb[i];
}

// Column statistics: sum2[c] += sum, sum2[cols+c] += sumsq.  blockDim.x == cols.
#define ROWS_PER_BLK 128
__global__ void k_colstats(const float* __restrict__ X, int rows, int cols,
                           float* __restrict__ sum2) {
  int c = threadIdx.x;
  int r0 = blockIdx.x * ROWS_PER_BLK;
  int rend = r0 + ROWS_PER_BLK; if (rend > rows) rend = rows;
  float s = 0.0f, ss = 0.0f;
  for (int r = r0; r < rend; ++r) {
    float v = X[(size_t)r * cols + c];
    s += v; ss += v * v;
  }
  atomAddF(&sum2[c], s);
  atomAddF(&sum2[cols + c], ss);
}

// BatchNorm (biased var) + optional ReLU:  Y = (X-m)*rsqrt(v+eps)*g + b
__global__ __launch_bounds__(256)
void k_bn(const float* __restrict__ X, float* __restrict__ Y, int rows, int cols,
          const float* __restrict__ sum2, const float* __restrict__ g,
          const float* __restrict__ b, int relu) {
  long i = (long)blockIdx.x * blockDim.x + threadIdx.x;
  if (i >= (long)rows * cols) return;
  int c = (int)(i % cols);
  float inv = 1.0f / (float)rows;
  float mean = sum2[c] * inv;
  float var  = sum2[cols + c] * inv - mean * mean;
  float y = (X[i] - mean) * rsqrtf(var + BN_EPS) * g[c] + b[c];
  if (relu) y = y > 0.0f ? y : 0.0f;
  Y[i] = y;
}

// Generic H-wide scatter-add: Y[to[p],c] += X[from[p],c]
__global__ __launch_bounds__(256)
void k_scatter(const float* __restrict__ X, const int* __restrict__ from,
               const int* __restrict__ to, float* __restrict__ Y, int Pn) {
  long i = (long)blockIdx.x * blockDim.x + threadIdx.x;
  if (i >= (long)Pn * HDIM) return;
  int p = (int)(i >> 7), c = (int)(i & 127);
  atomAddF(&Y[(size_t)to[p] * HDIM + c], X[(size_t)from[p] * HDIM + c]);
}

// xc += w_g2t * relu(u)
__global__ __launch_bounds__(256)
void k_xc_update(float* __restrict__ xc, const float* __restrict__ u,
                 const float* __restrict__ w, long n) {
  long i = (long)blockIdx.x * blockDim.x + threadIdx.x;
  if (i >= n) return;
  float r = u[i] > 0.0f ? u[i] : 0.0f;
  xc[i] += w[0] * r;
}

// h += w_t2g * leaky(v)
__global__ __launch_bounds__(256)
void k_h_update(float* __restrict__ h, const float* __restrict__ v,
                const float* __restrict__ w, long n) {
  long i = (long)blockIdx.x * blockDim.x + threadIdx.x;
  if (i >= n) return;
  float x = v[i];
  float l = x >= 0.0f ? x : 0.01f * x;
  h[i] += w[0] * l;
}

// readout[batch[n],c] += h[n,c]
__global__ __launch_bounds__(256)
void k_readout(const float* __restrict__ h, const int* __restrict__ batch,
               float* __restrict__ out) {
  long i = (long)blockIdx.x * blockDim.x + threadIdx.x;
  if (i >= (long)NATOMS * HDIM) return;
  int n = (int)(i >> 7), c = (int)(i & 127);
  atomAddF(&out[(size_t)batch[n] * HDIM + c], h[i]);
}

static inline unsigned gblk(long total, int blk) { return (unsigned)((total + blk - 1) / blk); }

extern "C" void kernel_launch(void* const* d_in, const int* in_sizes, int n_in,
                              void* d_out, int out_size, void* d_ws, size_t ws_size,
                              hipStream_t stream) {
  // ---- inputs --------------------------------------------------------------
  const int*   x        = (const int*)d_in[0];
  const int*   x_clique = (const int*)d_in[1];
  const int*   eidx     = (const int*)d_in[2];    // src = eidx, dst = eidx+E
  const int*   eattr    = (const int*)d_in[3];
  const int*   a2c_row  = (const int*)d_in[4];
  const int*   a2c_col  = (const int*)d_in[5];
  const int*   batch    = (const int*)d_in[6];
  const float* atom_emb = (const float*)d_in[8];
  const float* cl_emb   = (const float*)d_in[9];
  const float* cl_W     = (const float*)d_in[10];
  const float* cl_b     = (const float*)d_in[11];
  const float* bond_emb = (const float*)d_in[12];
  const float* gW1      = (const float*)d_in[13];
  const float* gb1      = (const float*)d_in[14];
  const float* bn_g     = (const float*)d_in[15];
  const float* bn_b     = (const float*)d_in[16];
  const float* gW2      = (const float*)d_in[17];
  const float* gb2      = (const float*)d_in[18];
  const float* geps     = (const float*)d_in[19];
  const float* gn_g     = (const float*)d_in[20];
  const float* gn_b     = (const float*)d_in[21];
  const float* sn_g     = (const float*)d_in[22];
  const float* sn_b     = (const float*)d_in[23];
  const float* aW       = (const float*)d_in[24];
  const float* ab       = (const float*)d_in[25];
  const float* cW       = (const float*)d_in[26];
  const float* cb       = (const float*)d_in[27];
  const float* w_g2t    = (const float*)d_in[28];
  const float* w_t2g    = (const float*)d_in[29];

  const int H = HDIM, H2 = H2DIM;
  const long NH = (long)NATOMS * H, NCH = (long)NCLIQ * H;

  // ---- workspace layout ----------------------------------------------------
  _Float16* wt_clq = (_Float16*)d_ws;             // [H][H]
  _Float16* wt1    = wt_clq + H * H;              // 3 x [2H][H]
  _Float16* wt2    = wt1 + 3 * H2 * H;            // 3 x [H][2H]
  _Float16* wta    = wt2 + 3 * H * H2;            // 3 x [H][H]
  _Float16* wtc    = wta + 3 * H * H;             // 3 x [H][H]
  float* fbase = (float*)(wtc + 3 * H * H);
  float* h   = fbase;                             // [N,H]
  float* zb  = h  + NH;                           // [N,H]   (agg/z, h2, u, clique gather)
  float* y1  = zb + NH;                           // [N,2H]  (also t | v)
  float* s   = y1 + (long)NATOMS * H2;            // [NC,H]
  float* xc  = s  + NCH;                          // [NC,H]
  float* st  = xc + NCH;                          // [512] col stats
  float* t   = y1;                                // alias
  float* vv  = y1 + NH;                           // alias

  float* out_xc = (float*)d_out;                  // [NC,H]
  float* out_ge = out_xc + NCH;                   // [N,H]
  float* out_ro = out_ge + NH;                    // [B,H]

  // ---- weight convert/transpose -------------------------------------------
  k_wT<<<gblk((long)H * H, 256), 256, 0, stream>>>(cl_W, wt_clq, H, H);
  for (int i = 0; i < NLAYER; ++i) {
    k_wT<<<gblk((long)H * H2, 256), 256, 0, stream>>>(gW1 + (long)i * H * H2, wt1 + (long)i * H2 * H, H,  H2);
    k_wT<<<gblk((long)H2 * H, 256), 256, 0, stream>>>(gW2 + (long)i * H2 * H, wt2 + (long)i * H * H2, H2, H);
    k_wT<<<gblk((long)H * H,  256), 256, 0, stream>>>(aW  + (long)i * H * H,  wta + (long)i * H * H,  H,  H);
    k_wT<<<gblk((long)H * H,  256), 256, 0, stream>>>(cW  + (long)i * H * H,  wtc + (long)i * H * H,  H,  H);
  }

  // ---- encoders ------------------------------------------------------------
  k_atom_enc<<<gblk(NH, 256), 256, 0, stream>>>(x, atom_emb, h, out_ge);
  k_clq_gather<<<gblk(NCH, 256), 256, 0, stream>>>(x_clique, cl_emb, zb);
  {
    int tiles = (NCLIQ / 16) * (H / 64);
    k_gemm_wmma<<<(tiles + 3) / 4, 128, 0, stream>>>(zb, wt_clq, cl_b, xc, NCLIQ, H, H);
  }

  // ---- layers --------------------------------------------------------------
  for (int i = 0; i < NLAYER; ++i) {
    // agg = scatter_dst(relu(h[src] + bond_enc))
    k_fill0<<<gblk(NH, 256), 256, 0, stream>>>(zb, NH);
    k_edge<<<gblk((long)NEDGES * H, 256), 256, 0, stream>>>(
        h, bond_emb + (long)i * 3 * VB_SZ * H, eattr, eidx, eidx + NEDGES, zb);
    // z = (1+eps)*h + agg
    k_make_z<<<gblk(NH, 256), 256, 0, stream>>>(h, zb, geps, i, NH);
    // y1 = z @ W1 + b1 -> BN -> relu   [N,2H]
    {
      int tiles = (NATOMS / 16) * (H2 / 64);
      k_gemm_wmma<<<(tiles + 3) / 4, 128, 0, stream>>>(zb, wt1 + (long)i * H2 * H, gb1 + (long)i * H2, y1, NATOMS, H, H2);
    }
    k_fill0<<<2, 256, 0, stream>>>(st, 512);
    k_colstats<<<gblk(NATOMS, ROWS_PER_BLK), H2, 0, stream>>>(y1, NATOMS, H2, st);
    k_bn<<<gblk((long)NATOMS * H2, 256), 256, 0, stream>>>(y1, y1, NATOMS, H2, st, bn_g + (long)i * H2, bn_b + (long)i * H2, 1);
    // h2 = y1 @ W2 + b2 -> BN(gn) -> relu -> h
    {
      int tiles = (NATOMS / 16) * (H / 64);
      k_gemm_wmma<<<(tiles + 3) / 4, 128, 0, stream>>>(y1, wt2 + (long)i * H * H2, gb2 + (long)i * H, zb, NATOMS, H2, H);
    }
    k_fill0<<<2, 256, 0, stream>>>(st, 512);
    k_colstats<<<gblk(NATOMS, ROWS_PER_BLK), H, 0, stream>>>(zb, NATOMS, H, st);
    k_bn<<<gblk(NH, 256), 256, 0, stream>>>(zb, h, NATOMS, H, st, gn_g + (long)i * H, gn_b + (long)i * H, 1);
    // s = scatter a2c ; u = s @ a2c_W + b ; xc += w_g2t*relu(u) ; xc = BN(sn)
    k_fill0<<<gblk(NCH, 256), 256, 0, stream>>>(s, NCH);
    k_scatter<<<gblk((long)NPAIR * H, 256), 256, 0, stream>>>(h, a2c_row, a2c_col, s, NPAIR);
    {
      int tiles = (NCLIQ / 16) * (H / 64);
      k_gemm_wmma<<<(tiles + 3) / 4, 128, 0, stream>>>(s, wta + (long)i * H * H, ab + (long)i * H, zb, NCLIQ, H, H);
    }
    k_xc_update<<<gblk(NCH, 256), 256, 0, stream>>>(xc, zb, w_g2t, NCH);
    k_fill0<<<2, 256, 0, stream>>>(st, 512);
    k_colstats<<<gblk(NCLIQ, ROWS_PER_BLK), H, 0, stream>>>(xc, NCLIQ, H, st);
    k_bn<<<gblk(NCH, 256), 256, 0, stream>>>(xc, xc, NCLIQ, H, st, sn_g + (long)i * H, sn_b + (long)i * H, 0);
    // t = scatter c2a ; v = t @ c2a_W + b ; h += w_t2g*leaky(v)
    k_fill0<<<gblk(NH, 256), 256, 0, stream>>>(t, NH);
    k_scatter<<<gblk((long)NPAIR * H, 256), 256, 0, stream>>>(xc, a2c_col, a2c_row, t, NPAIR);
    {
      int tiles = (NATOMS / 16) * (H / 64);
      k_gemm_wmma<<<(tiles + 3) / 4, 128, 0, stream>>>(t, wtc + (long)i * H * H, cb + (long)i * H, vv, NATOMS, H, H);
    }
    k_h_update<<<gblk(NH, 256), 256, 0, stream>>>(h, vv, w_t2g, NH);
  }

  // ---- outputs -------------------------------------------------------------
  k_fill0<<<gblk((long)NGRAPH * H, 256), 256, 0, stream>>>(out_ro, (long)NGRAPH * H);
  k_readout<<<gblk(NH, 256), 256, 0, stream>>>(h, batch, out_ro);
  k_copy<<<gblk(NCH, 256), 256, 0, stream>>>(xc, out_xc, NCH);
}
